// GraphNet3_11819749998954
// MI455X (gfx1250) — matrix operand order repaired and checked by reading, not compiled, not run on hardware
//
#include <hip/hip_runtime.h>
#include <math.h>

#define DEV __device__ __forceinline__

typedef __attribute__((ext_vector_type(16))) _Float16 v16h;
typedef __attribute__((ext_vector_type(8)))  float    v8f;

#define EPSN      1e-9f
#define NBR_NORM  0.22360679774997896f   // 1/sqrt(20)
#define DBOND     3.8f
#define DD        (DBOND*DBOND)

DEV float siluf(float x) { return x / (1.f + expf(-x)); }

DEV v8f wmma_f16(v16h a, v16h b, v8f c) {
  // D = A(16x32 f16) * B(32x16 f16) + C(16x16 f32)
  return __builtin_amdgcn_wmma_f32_16x16x32_f16(false, a, false, b, (short)0, c,
                                                false, false);
}

// GEMM for one 16-sample tile:  out[m][o] = sum_k cat[m][k] * Wt[o][k]
// cat[m] = [ x1(D1) | x2(D2) | outer(x1,x2)(D1*D2) ]  (bilinear folded into Wt)
// x1t: LDS [16*D1], x2t: LDS [16*D2], Wt: f16 [16][KPAD] (row-contiguous).
template<int D1, int D2, int KPAD>
DEV v8f mix_wmma(const float* x1t, const float* x2t, const _Float16* Wt, int lane) {
  v8f c = {};
  const int hi8 = (lane & 16) ? 8 : 0;
  const int m   = lane & 15;
  const float* r1 = x1t + m * D1;
  const float* r2 = x2t + m * D2;
#pragma unroll
  for (int kc = 0; kc < KPAD; kc += 32) {
    v16h a;
#pragma unroll
    for (int t = 0; t < 16; ++t) {
      int K = kc + 2 * hi8 + t;   // contiguous k order: k = 16h + t (matches B)
      float v = 0.f;
      if (K < D1)                      v = r1[K];
      else if (K < D1 + D2)            v = r2[K - D1];
      else if (K < D1 + D2 + D1 * D2) {
        int kz = K - D1 - D2;
        v = r1[kz / D2] * r2[kz % D2];
      }
      a[t] = (_Float16)v;
    }
    const v16h b = *(const v16h*)(Wt + (size_t)m * KPAD + kc + ((lane & 16) ? 16 : 0));
    c = wmma_f16(a, b, c);
  }
  return c;
}

DEV void store_tile(v8f c, const float* lb, float* out, int lane) {
  int n   = lane & 15;
  int hi8 = (lane & 16) ? 8 : 0;
  float b = lb[n];
#pragma unroll
  for (int r = 0; r < 8; ++r) out[(r + hi8) * 16 + n] = c[r] + b;
}

DEV void std_norm_row(float* row) {
  float m = 0.f;
  for (int i = 0; i < 16; ++i) m += row[i];
  m *= (1.f / 16.f);
  float v = 0.f;
  for (int i = 0; i < 16; ++i) { float d = row[i] - m; v += d * d; }
  float inv = 1.f / (sqrtf(v * (1.f / 15.f)) + EPSN);   // ddof=1
  for (int i = 0; i < 16; ++i) row[i] *= inv;
}

DEV float block_sum(float v, float* sh) {   // blockDim.x == 256
  int t = threadIdx.x;
  sh[t] = v; __syncthreads();
  for (int s = 128; s > 0; s >>= 1) { if (t < s) sh[t] += sh[t + s]; __syncthreads(); }
  float r = sh[0]; __syncthreads();
  return r;
}

// ---------------- small prep kernels ----------------

__global__ void k_zero(float* p, int n) {
  int i = blockIdx.x * blockDim.x + threadIdx.x;
  if (i < n) p[i] = 0.f;
}

// M = semi_unitary(proj), G = M M^T   (tiny, one thread)
__global__ void k_semi(const float* proj, float* M, float* G) {
  if (threadIdx.x != 0 || blockIdx.x != 0) return;
  float Mm[48];
  for (int i = 0; i < 48; ++i) Mm[i] = proj[i];
  for (int it = 0; it < 10; ++it) {
    float A[9];
    for (int r = 0; r < 3; ++r)
      for (int c = 0; c < 3; ++c) {
        float s = 0.f;
        for (int o = 0; o < 16; ++o) s += Mm[r*16+o] * Mm[c*16+o];
        A[r*3+c] = s - (r == c ? 1.f : 0.f);
      }
    float Nn[48];
    for (int r = 0; r < 3; ++r)
      for (int o = 0; o < 16; ++o)
        Nn[r*16+o] = Mm[r*16+o] -
          0.5f * (A[r*3+0]*Mm[o] + A[r*3+1]*Mm[16+o] + A[r*3+2]*Mm[32+o]);
    for (int i = 0; i < 48; ++i) Mm[i] = Nn[i];
  }
  for (int i = 0; i < 48; ++i) M[i] = Mm[i];
  for (int r = 0; r < 3; ++r)
    for (int c = 0; c < 3; ++c) {
      float s = 0.f;
      for (int o = 0; o < 16; ++o) s += Mm[r*16+o] * Mm[c*16+o];
      G[r*3+c] = s;
    }
}

// Fold bilinear+linear weights into f16 Wt[o][k], k = [x1 | x2 | x1⊗x2], zero-pad to KPAD.
template<int D1, int D2, int KPAD>
__global__ void k_fold(const float* bw, const float* lw, void* Wt_) {
  _Float16* Wt = (_Float16*)Wt_;
  int i = blockIdx.x * blockDim.x + threadIdx.x;
  if (i >= 16 * KPAD) return;
  int o = i / KPAD, k = i % KPAD;
  const int IN = 2 * D1 + D2;
  float v = 0.f;
  if (k < D1 + D2) v = lw[o * IN + k];
  else if (k < D1 + D2 + D1 * D2) {
    int kz = k - D1 - D2;
    int ii = kz / D2, jj = kz % D2;
    for (int o2 = 0; o2 < 16; ++o2)
      v += bw[(o2 * D1 + ii) * D2 + jj] * lw[o * IN + D1 + D2 + o2];
  }
  Wt[o * KPAD + k] = (_Float16)v;
}

__global__ void k_uplift(const float* R, const float* M, float* xn, int N) {
  int i = blockIdx.x * blockDim.x + threadIdx.x;
  if (i >= N * 16) return;
  int n = i >> 4, o = i & 15;
  xn[i] = R[n*3+0]*M[o] + R[n*3+1]*M[16+o] + R[n*3+2]*M[32+o];
}

__global__ void k_edge_attr(const float* R, const int* src, const int* dst,
                            float* ea, int E) {
  int e = blockIdx.x * blockDim.x + threadIdx.x;
  if (e >= E) return;
  int s = src[e], d = dst[e];
  float dx = R[d*3+0] - R[s*3+0];
  float dy = R[d*3+1] - R[s*3+1];
  float dz = R[d*3+2] - R[s*3+2];
  ea[e] = 1.f / (sqrtf(dx*dx + dy*dy + dz*dz) + EPSN);
}

// ---------------- WMMA block kernels (one 16-sample tile per 32-thread block) --------

__global__ void k_node_mix(const float* xn, const float* at, const void* Wt_,
                           const float* lb, float* out) {
  __shared__ float x1T[256], x2T[256], oT[256];
  int lane = threadIdx.x;
  int base = blockIdx.x * 256;   // 16 nodes * 16 feats
#pragma unroll
  for (int q = 0; q < 8; ++q) { int e = lane * 8 + q; x1T[e] = xn[base + e]; }
#pragma unroll
  for (int q = 0; q < 8; ++q) { int e = lane * 8 + q; x2T[e] = at[base + e]; }
  __syncthreads();
  v8f c = mix_wmma<16, 16, 288>(x1T, x2T, (const _Float16*)Wt_, lane);
  store_tile(c, lb, oT, lane);
  __syncthreads();
  if (lane < 16) {
    float* row = oT + lane * 16;
    std_norm_row(row);
    for (int i = 0; i < 16; ++i) out[base + lane * 16 + i] = row[i];
  }
}

__global__ void k_edges(const float* xnA, const float* ea,
                        const int* src, const int* dst,
                        const float* fc1w, const float* fc1b,
                        const float* fc2w, const float* fc2b,
                        const void* WtN2E, const float* lbN2E,
                        const void* WtMXE, const float* lbMXE,
                        float* accD, float* accS) {
  __shared__ float gT[256], aT[256], xeT[256], oT[256], eaT[16];
  int lane = threadIdx.x;
  int e0 = blockIdx.x * 16;
  if (lane < 16) {
    int e = e0 + lane;
    int s = src[e], d = dst[e];
    float att = ea[e];
    eaT[lane] = att;
#pragma unroll
    for (int i = 0; i < 16; ++i) {
      float xs = xnA[s*16+i], xd = xnA[d*16+i];
      float w  = siluf(fc1w[i] * att + fc1b[i]);
      gT[lane*16+i] = w * (xs - xd);
      aT[lane*16+i] = w * (xs + xd) * 0.5f;
    }
  }
  __syncthreads();
  v8f c1 = mix_wmma<16, 16, 288>(gT, aT, (const _Float16*)WtN2E, lane);
  store_tile(c1, lbN2E, xeT, lane);
  __syncthreads();
  v8f c2 = mix_wmma<16, 1, 64>(xeT, eaT, (const _Float16*)WtMXE, lane);
  store_tile(c2, lbMXE, oT, lane);
  __syncthreads();
  if (lane < 16) {
    int e = e0 + lane;
    int s = src[e], d = dst[e];
    float att = ea[e];
    float* row = oT + lane * 16;
    std_norm_row(row);
#pragma unroll
    for (int i = 0; i < 16; ++i) {
      float w2 = siluf(fc2w[i] * att + fc2b[i]);
      float we = w2 * row[i];
      unsafeAtomicAdd(&accD[d*16+i], we);   // x1 = segment_sum over dst
      unsafeAtomicAdd(&accS[s*16+i], we);   // x2 = segment_sum over src
    }
  }
}

__global__ void k_e2n(const float* accD, const float* accS, const void* Wt_,
                      const float* lb, const float* angw, const float* angb,
                      float* xn /* in: xn_org, out: combined */) {
  __shared__ float x1T[256], x2T[256], oT[256];
  int lane = threadIdx.x;
  int base = blockIdx.x * 256;
#pragma unroll
  for (int q = 0; q < 8; ++q) {
    int e = lane * 8 + q;
    float a = accD[base + e] * NBR_NORM;
    float b = accS[base + e] * NBR_NORM;
    x1T[e] = a - b;
    x2T[e] = a + b;
  }
  __syncthreads();
  v8f c = mix_wmma<16, 16, 288>(x1T, x2T, (const _Float16*)Wt_, lane);
  store_tile(c, lb, oT, lane);
  __syncthreads();
  if (lane < 16) {
    float* row = oT + lane * 16;
    for (int i = 0; i < 16; ++i) row[i] = siluf(row[i]);
    std_norm_row(row);
    float ang = angb[0];
    for (int i = 0; i < 16; ++i) ang += row[i] * angw[i];
    ang *= 0.1f;
    float ca = cosf(ang), sa = sinf(ang);
    for (int i = 0; i < 16; ++i) {
      int idx = base + lane * 16 + i;
      xn[idx] = ca * xn[idx] + sa * row[i];
    }
  }
}

// ---------------- constraint-projection kernels ----------------
// scal layout: [0]=cn2 [1]=lamnorm2 [2]=alpha [3]=a_step [4..15]=trial norms^2

__global__ void k_project(const float* xn, const float* M, float* R, int N) {
  int i = blockIdx.x * blockDim.x + threadIdx.x;
  if (i >= N * 3) return;
  int n = i / 3, c = i % 3;
  float s = 0.f;
  for (int o = 0; o < 16; ++o) s += xn[n*16+o] * M[c*16+o];
  R[i] = s;
}

__global__ void k_bonds(const float* X, float* cv, float* scal, int N) {
  __shared__ float sh[256];
  int i = blockIdx.x * 256 + threadIdx.x;
  float loc = 0.f;
  if (i < N - 1) {
    float dx = X[(i+1)*3+0] - X[i*3+0];
    float dy = X[(i+1)*3+1] - X[i*3+1];
    float dz = X[(i+1)*3+2] - X[i*3+2];
    float cc = dx*dx + dy*dy + dz*dz - DD;
    cv[i] = cc;
    loc = cc * cc;
  }
  float s = block_sum(loc, sh);
  if (threadIdx.x == 0) unsafeAtomicAdd(&scal[0], s);
}

__global__ void k_lamP(const float* X, const float* cv, const float* M, const float* G,
                       float* lam, float* P, float* scal, int N, int first) {
  __shared__ float sh[256];
  int n = blockIdx.x * 256 + threadIdx.x;
  float loc = 0.f;
  if (n < N) {
    float Dv[3];
    float cm = (n >= 1)     ? cv[n-1] : 0.f;
    float cn = (n < N - 1)  ? cv[n]   : 0.f;
    for (int c = 0; c < 3; ++c) {
      float Cm = (n >= 1)    ? cm * (X[n*3+c]     - X[(n-1)*3+c]) : 0.f;
      float Cn = (n < N - 1) ? cn * (X[(n+1)*3+c] - X[n*3+c])     : 0.f;
      Dv[c] = 2.f * (Cm - Cn);
    }
    for (int o = 0; o < 16; ++o) {
      float l = Dv[0]*M[o] + Dv[1]*M[16+o] + Dv[2]*M[32+o];
      lam[n*16+o] = l;
      loc += l * l;
    }
    for (int c = 0; c < 3; ++c)
      P[n*3+c] = G[c*3+0]*Dv[0] + G[c*3+1]*Dv[1] + G[c*3+2]*Dv[2];
  }
  if (first) {
    float s = block_sum(loc, sh);
    if (threadIdx.x == 0) unsafeAtomicAdd(&scal[1], s);
  }
}

__global__ void k_alpha(float* scal) {
  if (threadIdx.x == 0 && blockIdx.x == 0) scal[2] = 1.f / sqrtf(scal[1]);
}

// All 12 line-search trials in one pass: trials[l] = ||c(X - alpha*2^-l * P)||^2
__global__ void k_ls(const float* X, const float* P, float* scal, int N) {
  __shared__ float s12[12];
  int i = blockIdx.x * 256 + threadIdx.x;
  float loc[12];
#pragma unroll
  for (int l = 0; l < 12; ++l) loc[l] = 0.f;
  if (i < N - 1) {
    float dx[3], dp[3];
    for (int c = 0; c < 3; ++c) {
      dx[c] = X[(i+1)*3+c] - X[i*3+c];
      dp[c] = P[(i+1)*3+c] - P[i*3+c];
    }
    float a = scal[2];
#pragma unroll
    for (int l = 0; l < 12; ++l) {
      float v = 0.f;
      for (int c = 0; c < 3; ++c) { float t = dx[c] - a * dp[c]; v += t * t; }
      float ct = v - DD;
      loc[l] = ct * ct;
      a *= 0.5f;
    }
  }
  if (threadIdx.x < 12) s12[threadIdx.x] = 0.f;
  __syncthreads();
#pragma unroll
  for (int l = 0; l < 12; ++l) atomicAdd(&s12[l], loc[l]);
  __syncthreads();
  if (threadIdx.x < 12) unsafeAtomicAdd(&scal[4 + threadIdx.x], s12[threadIdx.x]);
}

__global__ void k_pick(float* scal) {
  if (threadIdx.x != 0 || blockIdx.x != 0) return;
  float cn = sqrtf(scal[0]);
  float a  = scal[2];
  int ls = 0;
  for (; ls <= 10; ++ls) {
    if (!(sqrtf(scal[4 + ls]) >= cn)) break;
    a *= 0.5f;
  }
  if (ls == 0) a *= 1.5f;
  scal[2] = a;            // carried alpha
  scal[3] = a;            // step actually taken
  scal[0] = 0.f; scal[1] = 0.f;
  for (int l = 0; l < 12; ++l) scal[4 + l] = 0.f;
}

__global__ void k_update(float* xn, const float* lam, float* X, const float* P,
                         const float* scal, int N) {
  int i = blockIdx.x * blockDim.x + threadIdx.x;
  float a = scal[3];
  if (i < N * 16) xn[i] -= a * lam[i];
  if (i < N * 3)  X[i]  -= a * P[i];
}

// ---------------- host driver ----------------

extern "C" void kernel_launch(void* const* d_in, const int* in_sizes, int n_in,
                              void* d_out, int out_size, void* d_ws, size_t ws_size,
                              hipStream_t stream) {
  (void)in_sizes; (void)n_in; (void)out_size; (void)ws_size;
  const int N = 50000, E = 800000, L = 2;

  // ---- inputs (setup_inputs dict order, depth-first over params) ----
  int id = 0;
  const float* input   = (const float*)d_in[id++];
  const float* xn_attr = (const float*)d_in[id++];
  const float* proj    = (const float*)d_in[id++];
  struct MixP { const float *bw, *lw, *lb; };
  struct LayP {
    const float *fc1w, *fc1b, *fc2w, *fc2b;
    MixP mxn, mxe, n2e, e2n;
    const float *angw, *angb;
  } lay[2];
  for (int l = 0; l < L; ++l) {
    lay[l].fc1w = (const float*)d_in[id++];
    lay[l].fc1b = (const float*)d_in[id++];
    lay[l].fc2w = (const float*)d_in[id++];
    lay[l].fc2b = (const float*)d_in[id++];
    MixP* ms[4] = { &lay[l].mxn, &lay[l].mxe, &lay[l].n2e, &lay[l].e2n };
    for (int m = 0; m < 4; ++m) {
      ms[m]->bw = (const float*)d_in[id++];
      ms[m]->lw = (const float*)d_in[id++];
      ms[m]->lb = (const float*)d_in[id++];
    }
    lay[l].angw = (const float*)d_in[id++];
    lay[l].angb = (const float*)d_in[id++];
  }
  const int* src = (const int*)d_in[id++];
  const int* dst = (const int*)d_in[id++];

  // ---- workspace carve (256B-aligned bump allocator) ----
  uintptr_t w = (uintptr_t)d_ws;
  auto take = [&w](size_t bytes) -> void* {
    uintptr_t p = w;
    w = (w + bytes + 255) & ~(uintptr_t)255;
    return (void*)p;
  };
  float* scal  = (float*)take(64 * 4);
  float* M     = (float*)take(48 * 4);
  float* G     = (float*)take(12 * 4);
  float* xn    = (float*)take((size_t)N * 16 * 4);
  float* xnA   = (float*)take((size_t)N * 16 * 4);
  float* accD  = (float*)take((size_t)N * 16 * 4 * 2);
  float* accS  = accD + (size_t)N * 16;
  float* xeat  = (float*)take((size_t)E * 4);
  float* X     = (float*)take((size_t)N * 3 * 4);
  float* P     = (float*)take((size_t)N * 3 * 4);
  float* lam   = (float*)take((size_t)N * 16 * 4);
  float* cvec  = (float*)take((size_t)N * 4);
  float* Rcur  = (float*)take((size_t)N * 3 * 4);
  void* WtMXN[2]; void* WtN2E[2]; void* WtE2N[2]; void* WtMXE[2];
  for (int l = 0; l < L; ++l) {
    WtMXN[l] = take(16 * 288 * 2);
    WtN2E[l] = take(16 * 288 * 2);
    WtE2N[l] = take(16 * 288 * 2);
    WtMXE[l] = take(16 * 64  * 2);
  }

  // ---- prep ----
  k_zero<<<1, 64, 0, stream>>>(scal, 64);
  k_semi<<<1, 32, 0, stream>>>(proj, M, G);
  for (int l = 0; l < L; ++l) {
    k_fold<16, 16, 288><<<(16*288 + 255)/256, 256, 0, stream>>>(lay[l].mxn.bw, lay[l].mxn.lw, WtMXN[l]);
    k_fold<16, 16, 288><<<(16*288 + 255)/256, 256, 0, stream>>>(lay[l].n2e.bw, lay[l].n2e.lw, WtN2E[l]);
    k_fold<16, 16, 288><<<(16*288 + 255)/256, 256, 0, stream>>>(lay[l].e2n.bw, lay[l].e2n.lw, WtE2N[l]);
    k_fold<16, 1, 64 ><<<(16*64  + 255)/256, 256, 0, stream>>>(lay[l].mxe.bw, lay[l].mxe.lw, WtMXE[l]);
  }
  k_uplift<<<(N*16 + 255)/256, 256, 0, stream>>>(input, M, xn, N);

  // ---- layers ----
  for (int l = 0; l < L; ++l) {
    const float* Rin = (l == 0) ? input : Rcur;
    k_edge_attr<<<(E + 255)/256, 256, 0, stream>>>(Rin, src, dst, xeat, E);
    k_zero<<<(2*N*16 + 255)/256, 256, 0, stream>>>(accD, 2*N*16);

    k_node_mix<<<N/16, 32, 0, stream>>>(xn, xn_attr, WtMXN[l], lay[l].mxn.lb, xnA);
    k_edges<<<E/16, 32, 0, stream>>>(xnA, xeat, src, dst,
                                     lay[l].fc1w, lay[l].fc1b, lay[l].fc2w, lay[l].fc2b,
                                     WtN2E[l], lay[l].n2e.lb,
                                     WtMXE[l], lay[l].mxe.lb,
                                     accD, accS);
    k_e2n<<<N/16, 32, 0, stream>>>(accD, accS, WtE2N[l], lay[l].e2n.lb,
                                   lay[l].angw, lay[l].angb, xn);

    // constraint projection: 100 steps, X maintained incrementally
    k_project<<<(N*3 + 255)/256, 256, 0, stream>>>(xn, M, X, N);
    for (int j = 0; j < 100; ++j) {
      k_bonds<<<(N - 1 + 255)/256, 256, 0, stream>>>(X, cvec, scal, N);
      k_lamP<<<(N + 255)/256, 256, 0, stream>>>(X, cvec, M, G, lam, P, scal, N, j == 0);
      if (j == 0) k_alpha<<<1, 32, 0, stream>>>(scal);
      k_ls<<<(N - 1 + 255)/256, 256, 0, stream>>>(X, P, scal, N);
      k_pick<<<1, 32, 0, stream>>>(scal);
      k_update<<<(N*16 + 255)/256, 256, 0, stream>>>(xn, lam, X, P, scal, N);
    }

    float* Rout = (l == L - 1) ? (float*)d_out : Rcur;
    k_project<<<(N*3 + 255)/256, 256, 0, stream>>>(xn, M, Rout, N);
  }
}
